// Pointnet2_66872640799427
// MI455X (gfx1250) — compile-verified, matrix operand order checked by reading, not run on hardware
//
#include <hip/hip_runtime.h>

typedef __attribute__((ext_vector_type(16))) _Float16 v16h;
typedef __attribute__((ext_vector_type(8)))  _Float16 v8h;
typedef __attribute__((ext_vector_type(8)))  float    v8f;

typedef int v4i __attribute__((vector_size(16)));
typedef v4i __attribute__((address_space(1)))* as1_v4i_ptr;   // global
typedef v4i __attribute__((address_space(3)))* as3_v4i_ptr;   // LDS

// ---- CDNA5 async global->LDS copy (ASYNCcnt path), with portable fallback ----
__device__ __forceinline__ void async_copy16(const _Float16* g, _Float16* l)
{
#if __has_builtin(__builtin_amdgcn_global_load_async_to_lds_b128)
    // AS1 ptr == generic VA; AS3 ptr == generic[31:0] (ISA 10.2 LDS aperture).
    __builtin_amdgcn_global_load_async_to_lds_b128(
        (as1_v4i_ptr)(unsigned long long)g,
        (as3_v4i_ptr)(unsigned)(unsigned long long)l,
        /*offset=*/0, /*cpol=*/0);
#else
    *(v8h*)l = *(const v8h*)g;
#endif
}

__device__ __forceinline__ void wait_async0()
{
#if __has_builtin(__builtin_amdgcn_global_load_async_to_lds_b128)
#if __has_builtin(__builtin_amdgcn_s_wait_asynccnt)
    __builtin_amdgcn_s_wait_asynccnt(0);
#else
    asm volatile("s_wait_asynccnt 0x0" ::: "memory");
#endif
#endif
}

// ---------------------------------------------------------------------------
// Farthest point sampling: one block per batch, distance array LDS-resident
// (32768 floats = 128 KB, legal under CDNA5's 320 KB/WGP LDS).
// pts element (b,n,c) at pts[b*sb + n*sn + c*sc].
// ---------------------------------------------------------------------------
__global__ __launch_bounds__(1024)
void fps_kernel(const float* __restrict__ pts, int N, int npoint,
                long long sb, long long sn, long long sc,
                int* __restrict__ out_idx, float* __restrict__ out_xyz)
{
    extern __shared__ float smem[];
    float* dist = smem;                       // N floats
    float* rval = smem + N;                   // blockDim floats
    int*   ridx = (int*)(rval + blockDim.x);  // blockDim ints
    __shared__ float cent[3];
    __shared__ int   s_far;

    const int b = blockIdx.x, tid = threadIdx.x, T = blockDim.x;
    const float* base = pts + (size_t)b * sb;

    for (int n = tid; n < N; n += T) dist[n] = 1e10f;
    if (tid == 0) s_far = 0;
    __syncthreads();

    for (int it = 0; it < npoint; ++it) {
        if (tid == 0) {
            int far = s_far;
            float cx = base[(size_t)far * sn];
            float cy = base[(size_t)far * sn + sc];
            float cz = base[(size_t)far * sn + 2 * sc];
            cent[0] = cx; cent[1] = cy; cent[2] = cz;
            out_idx[b * npoint + it] = far;
            float* o = out_xyz + ((size_t)b * npoint + it) * 3;
            o[0] = cx; o[1] = cy; o[2] = cz;
        }
        __syncthreads();
        float cx = cent[0], cy = cent[1], cz = cent[2];
        float bv = -1.0f; int bi = 0;
        for (int n = tid; n < N; n += T) {
            float dx = base[(size_t)n * sn]          - cx;
            float dy = base[(size_t)n * sn + sc]     - cy;
            float dz = base[(size_t)n * sn + 2 * sc] - cz;
            float d  = dx * dx + dy * dy + dz * dz;
            float dd = dist[n];
            dd = d < dd ? d : dd;
            dist[n] = dd;
            if (dd > bv) { bv = dd; bi = n; }   // first-max within thread
        }
        rval[tid] = bv; ridx[tid] = bi;
        __syncthreads();
        for (int s = T >> 1; s > 0; s >>= 1) {
            if (tid < s) {
                float v2 = rval[tid + s]; int i2 = ridx[tid + s];
                // argmax tie-break: lowest index wins (matches jnp.argmax)
                if (v2 > rval[tid] || (v2 == rval[tid] && i2 < ridx[tid])) {
                    rval[tid] = v2; ridx[tid] = i2;
                }
            }
            __syncthreads();
        }
        if (tid == 0) s_far = ridx[0];
        __syncthreads();
    }
}

// ---------------------------------------------------------------------------
// Ball query + grouping, fused. One wave32 per query point. The reference's
// "mark OOR as N, sort, take first nsample, pad with first" is equivalent to
// collecting the first nsample in-radius indices in ascending index order.
// Writes grouped features [rel_xyz(3) | point_feats(pC) | zeros..Cp] as f16,
// rows padded to Cp (multiple of 32) so the GEMM K-loop needs no bounds checks.
// ---------------------------------------------------------------------------
__global__ __launch_bounds__(256)
void ballquery_group_kernel(const float* __restrict__ pts,
                            long long sb, long long sn, long long sc, int N,
                            const float* __restrict__ new_xyz,  // (Q,3) row-major
                            int S, int nsample, float r2,
                            const float* __restrict__ featf32,  // strided, or null
                            long long fb, long long fn, long long fc,
                            const _Float16* __restrict__ featf16, // (B,N,pC) or null
                            int pC, int Cp,
                            _Float16* __restrict__ out,          // (Q, nsample, Cp)
                            int Q)
{
    __shared__ int slots[8][64];
    const int wid  = threadIdx.x >> 5;
    const int lane = threadIdx.x & 31;
    const int q = blockIdx.x * 8 + wid;
    if (q >= Q) return;                 // uniform per-wave
    const int b = q / S;
    const float* base = pts + (size_t)b * sb;
    const float qx = new_xyz[q * 3], qy = new_xyz[q * 3 + 1], qz = new_xyz[q * 3 + 2];

    int cnt = 0;
    for (int n0 = 0; n0 < N && cnt < nsample; n0 += 32) {
        int n = n0 + lane;
        bool ok = false;
        if (n < N) {
            float dx = base[(size_t)n * sn]          - qx;
            float dy = base[(size_t)n * sn + sc]     - qy;
            float dz = base[(size_t)n * sn + 2 * sc] - qz;
            ok = (dx * dx + dy * dy + dz * dz) <= r2;
        }
        unsigned m = (unsigned)__ballot(ok);
        if (ok) {
            int pos = cnt + __popc(m & ((1u << lane) - 1u));
            if (pos < nsample) slots[wid][pos] = n;
        }
        cnt += __popc(m);
    }
    if (cnt > nsample) cnt = nsample;
    __syncthreads();
    // pad with first in-radius index (query centroid itself guarantees cnt>=1)
    for (int j = cnt + lane; j < nsample; j += 32) slots[wid][j] = slots[wid][0];
    __syncthreads();

    const int C = 3 + pC;
    for (int j = lane; j < nsample; j += 32) {
        int n = slots[wid][j];
        _Float16* orow = out + ((size_t)q * nsample + j) * Cp;
        orow[0] = (_Float16)(base[(size_t)n * sn]          - qx);
        orow[1] = (_Float16)(base[(size_t)n * sn + sc]     - qy);
        orow[2] = (_Float16)(base[(size_t)n * sn + 2 * sc] - qz);
        if (featf16) {
            const _Float16* f = featf16 + ((size_t)b * N + n) * pC;
            for (int c = 0; c < pC; ++c) orow[3 + c] = f[c];
        } else if (featf32) {
            const float* f = featf32 + (size_t)b * fb + (size_t)n * fn;
            for (int c = 0; c < pC; ++c) orow[3 + c] = (_Float16)f[(size_t)c * fc];
        }
        for (int c = C; c < Cp; ++c) orow[c] = (_Float16)0.0f;   // K-pad
    }
}

// ---------------------------------------------------------------------------
// Pack weights (O x C, f32, row-major) into WMMA B-fragment order, f16:
// wpack[((tO*nK + ks)*32 + lane)*16 + h]  with the ISA 7.12.2 16-bit layout,
// zero-filled for o>=O or c>=C. One contiguous 32B v16h per (tile,ks,lane).
// ---------------------------------------------------------------------------
__global__ __launch_bounds__(256)
void pack_weights_kernel(const float* __restrict__ W, int O, int C,
                         int nK, int tilesO, _Float16* __restrict__ wpack)
{
    int i = blockIdx.x * blockDim.x + threadIdx.x;
    int total = tilesO * nK * 512;
    if (i >= total) return;
    int h    = i & 15;
    int lane = (i >> 4) & 31;
    int rest = i >> 9;            // tO*nK + ks
    int ks   = rest % nK;
    int tO   = rest / nK;
    int ii = h >> 1, j = h & 1;
    int khi = (lane & 16) ? 8 : 0;
    int kk = (ii < 4) ? (khi + 2 * ii + j) : (16 + khi + 2 * (ii - 4) + j);
    int c = ks * 32 + kk;
    int o = tO * 16 + (lane & 15);
    _Float16 v = (_Float16)0.0f;
    if (o < O && c < C) v = (_Float16)W[(size_t)o * C + c];
    wpack[i] = v;
}

// ---------------------------------------------------------------------------
// GEMM + bias + ReLU via V_WMMA_F32_16X16X32_F16 (f32 accumulate).
// Block = 8 waves = 8 consecutive M-tiles x ONE O-tile-pair. The pair's
// pre-packed B fragments (2*nK KB) are staged once into LDS with the CDNA5
// async global->LDS path (ASYNCcnt + s_wait_asynccnt + barrier); the K-loop
// then mixes global b128 A loads (LOADcnt) with LDS b128 B reads (DScnt) and
// two back-to-back WMMAs per step. No bounds checks in the K-loop; output is
// written with stride Op and zero-filled for col>=O (keeps K-padding zero).
// ---------------------------------------------------------------------------
__global__ __launch_bounds__(256)
void gemm_relu_kernel(const _Float16* __restrict__ A,
                      const _Float16* __restrict__ wpack,
                      const float* __restrict__ bias,
                      _Float16* __restrict__ out,
                      int Cp, int O, int Op, int tilePairs, int nK)
{
    __shared__ _Float16 sB[2 * 5 * 512];     // supports nK <= 5 (10 KB)

    const int wid  = threadIdx.x >> 5;
    const int lane = threadIdx.x & 31;
    const int tg   = blockIdx.x % tilePairs;
    const int tM   = (blockIdx.x / tilePairs) * 8 + wid;

    // ---- stage this block's B fragments into LDS (async copy) ----
    {
        const int nchunk = (2 * nK * 512) / 8;                 // 16-byte chunks
        const _Float16* gsrc = wpack + (size_t)(2 * tg) * nK * 512;
        for (int e = threadIdx.x; e < nchunk; e += 256)
            async_copy16(gsrc + e * 8, &sB[e * 8]);
        wait_async0();
        __syncthreads();
    }

    const int row  = (tM << 4) + (lane & 15);
    const int col0 = (tg << 5) + (lane & 15);
    const int col1 = col0 + 16;
    const int khi  = (lane & 16) ? 8 : 0;

    const _Float16* arow = A + (size_t)row * Cp + khi;
    const _Float16* sb0  = &sB[lane * 16];
    const _Float16* sb1  = &sB[(size_t)nK * 512 + lane * 16];

    v8f acc0 = {}, acc1 = {};
    for (int ks = 0; ks < nK; ++ks) {
        const _Float16* pa = arow + ks * 32;
        v8h lo = *(const v8h*)(pa);           // halves 0..7  : K = khi..khi+7
        v8h hi = *(const v8h*)(pa + 16);      // halves 8..15 : K = 16+khi..23+khi
        v16h a = __builtin_shufflevector(lo, hi, 0, 1, 2, 3, 4, 5, 6, 7,
                                                 8, 9, 10, 11, 12, 13, 14, 15);
        v16h b0 = *(const v16h*)(sb0 + (size_t)ks * 512);   // ds_load_b128 x2
        v16h b1 = *(const v16h*)(sb1 + (size_t)ks * 512);
        acc0 = __builtin_amdgcn_wmma_f32_16x16x32_f16(false, a, false, b0,
                                                      (short)0, acc0, false, false);
        acc1 = __builtin_amdgcn_wmma_f32_16x16x32_f16(false, a, false, b1,
                                                      (short)0, acc1, false, false);
    }

    const float bv0 = (col0 < O) ? bias[col0] : 0.0f;
    const float bv1 = (col1 < O) ? bias[col1] : 0.0f;
    const int mbase = (tM << 4) + ((lane >> 4) << 3);   // D: VGPR v -> M=v (+8 hi lanes)
    #pragma unroll
    for (int v = 0; v < 8; ++v) {
        float r0 = (col0 < O) ? (acc0[v] + bv0) : 0.0f;
        float r1 = (col1 < O) ? (acc1[v] + bv1) : 0.0f;
        r0 = r0 > 0.0f ? r0 : 0.0f;
        r1 = r1 > 0.0f ? r1 : 0.0f;
        _Float16* orow = out + (size_t)(mbase + v) * Op;
        orow[col0] = (_Float16)r0;
        orow[col1] = (_Float16)r1;
    }
}

// ---------------------------------------------------------------------------
// Max pool over the neighbor axis: (Q,K,Cs) f16 -> (Q,C) f16 or f32.
// ---------------------------------------------------------------------------
__global__ __launch_bounds__(256)
void maxpool_kernel(const _Float16* __restrict__ in, int Q, int K, int C, int Cs,
                    _Float16* __restrict__ out16, float* __restrict__ out32)
{
    int i = blockIdx.x * blockDim.x + threadIdx.x;
    if (i >= Q * C) return;
    int q = i / C, c = i - q * C;
    const _Float16* p = in + (size_t)q * K * Cs + c;
    float m = -1e30f;
    for (int k = 0; k < K; ++k) {
        float v = (float)p[(size_t)k * Cs];
        m = v > m ? v : m;
    }
    if (out32) out32[i] = m; else out16[i] = (_Float16)m;
}

// (B,S,3) -> (B,3,S) for the coordinate output.
__global__ __launch_bounds__(256)
void xyz_transpose_kernel(const float* __restrict__ in, float* __restrict__ out,
                          int B, int S)
{
    int i = blockIdx.x * blockDim.x + threadIdx.x;
    if (i >= B * S * 3) return;
    int b = i / (S * 3);
    int r = i - b * S * 3;
    int c = r / S;
    int s = r - c * S;
    out[i] = in[((size_t)b * S + s) * 3 + c];
}

// ---------------------------------------------------------------------------
extern "C" void kernel_launch(void* const* d_in, const int* in_sizes, int n_in,
                              void* d_out, int out_size, void* d_ws, size_t ws_size,
                              hipStream_t stream)
{
    (void)in_sizes; (void)n_in; (void)out_size; (void)ws_size;
    const float* xyz = (const float*)d_in[0];
    const float* w1a = (const float*)d_in[1];  const float* b1a = (const float*)d_in[2];
    const float* w1b = (const float*)d_in[3];  const float* b1b = (const float*)d_in[4];
    const float* w1c = (const float*)d_in[5];  const float* b1c = (const float*)d_in[6];
    const float* w2a = (const float*)d_in[7];  const float* b2a = (const float*)d_in[8];
    const float* w2b = (const float*)d_in[9];  const float* b2b = (const float*)d_in[10];
    const float* w2c = (const float*)d_in[11]; const float* b2c = (const float*)d_in[12];

    const int B = 8, N1 = 32768, S1 = 512, K1 = 32, S2 = 256, K2 = 64;
    const int M = B * S1 * K1;   // == B*S2*K2 == 131072

    // workspace carve-up
    char* ws = (char*)d_ws;
    size_t off = 0;
    auto alloc = [&](size_t bytes) -> void* {
        bytes = (bytes + 255) & ~(size_t)255;
        void* p = ws + off; off += bytes; return p;
    };
    int*      idx1  = (int*)     alloc((size_t)B * S1 * sizeof(int));
    float*    nxyz1 = (float*)   alloc((size_t)B * S1 * 3 * sizeof(float));
    int*      idx2  = (int*)     alloc((size_t)B * S2 * sizeof(int));
    float*    nxyz2 = (float*)   alloc((size_t)B * S2 * 3 * sizeof(float));
    _Float16* l1f   = (_Float16*)alloc((size_t)B * S1 * 128 * sizeof(_Float16));
    _Float16* wpack = (_Float16*)alloc((size_t)18 * 5 * 512 * sizeof(_Float16)); // max tilesO*nK*512
    _Float16* bufA  = (_Float16*)alloc((size_t)M * 160 * sizeof(_Float16));
    _Float16* bufB  = (_Float16*)alloc((size_t)M * 288 * sizeof(_Float16));

    auto gemm = [&](const _Float16* A, int Cp, const float* W, const float* bias,
                    int C, _Float16* out, int O, int Op) {
        int tilesO = Op / 16;          // Op is a multiple of 32 => tilesO even
        int nK = Cp / 32;
        int total = tilesO * nK * 512;
        pack_weights_kernel<<<(total + 255) / 256, 256, 0, stream>>>(
            W, O, C, nK, tilesO, wpack);
        int tilePairs = tilesO / 2;
        int blocks = (M / 16 / 8) * tilePairs;   // 8 M-tiles per block, exact
        gemm_relu_kernel<<<blocks, 256, 0, stream>>>(
            A, wpack, bias, out, Cp, O, Op, tilePairs, nK);
    };

    // ---------------- SA1 ----------------
    {
        size_t sm = (size_t)N1 * 4 + 1024 * 8;
        fps_kernel<<<B, 1024, sm, stream>>>(xyz, N1, S1,
                                            (long long)6 * N1, 1, N1, idx1, nxyz1);
    }
    ballquery_group_kernel<<<(B * S1) / 8, 256, 0, stream>>>(
        xyz, (long long)6 * N1, 1, N1, N1,
        nxyz1, S1, K1, 0.025f * 0.025f,
        /*featf32=*/xyz + (size_t)3 * N1, (long long)6 * N1, 1, N1,
        /*featf16=*/(const _Float16*)nullptr, /*pC=*/3, /*Cp=*/32,
        bufA, B * S1);
    gemm(bufA, 32,  w1a, b1a,   6, bufB,  64,  64);
    gemm(bufB, 64,  w1b, b1b,  64, bufA,  64,  64);
    gemm(bufA, 64,  w1c, b1c,  64, bufB, 128, 128);
    maxpool_kernel<<<(B * S1 * 128 + 255) / 256, 256, 0, stream>>>(
        bufB, B * S1, K1, 128, 128, l1f, nullptr);

    // ---------------- SA2 ----------------
    {
        size_t sm = (size_t)S1 * 4 + 1024 * 8;
        fps_kernel<<<B, 1024, sm, stream>>>(nxyz1, S1, S2,
                                            (long long)S1 * 3, 3, 1, idx2, nxyz2);
    }
    ballquery_group_kernel<<<(B * S2) / 8, 256, 0, stream>>>(
        nxyz1, (long long)S1 * 3, 3, 1, S1,
        nxyz2, S2, K2, 0.05f * 0.05f,
        /*featf32=*/(const float*)nullptr, 0, 0, 0,
        /*featf16=*/l1f, /*pC=*/128, /*Cp=*/160,
        bufA, B * S2);
    gemm(bufA, 160, w2a, b2a, 131, bufB, 128, 128);
    gemm(bufB, 128, w2b, b2b, 128, bufA, 128, 128);
    gemm(bufA, 128, w2c, b2c, 128, bufB, 285, 288);

    float* out_pts = (float*)d_out + (size_t)B * 3 * S2;   // second tuple element
    maxpool_kernel<<<(B * S2 * 285 + 255) / 256, 256, 0, stream>>>(
        bufB, B * S2, K2, 285, 288, nullptr, out_pts);
    xyz_transpose_kernel<<<(B * S2 * 3 + 255) / 256, 256, 0, stream>>>(
        nxyz2, (float*)d_out, B, S2);
}
